// GAT_44641890075194
// MI455X (gfx1250) — matrix operand order repaired and checked by reading, not compiled, not run on hardware
//
#include <hip/hip_runtime.h>
#include <hip/hip_bf16.h>

typedef _Float16 v16h __attribute__((ext_vector_type(16)));
typedef float    v8f  __attribute__((ext_vector_type(8)));

#define N_NODES  50000
#define N_EDGES  800000
#define HIDDEN   128
#define HEADS    8
#define NEG_SLOPE 0.01f
#define N_TILES  (N_NODES / 16)   // 50000/16 = 3125 exactly

// ---------------------------------------------------------------------------
// Zero-fill (out and denom are accumulated with atomics; harness poisons them)
// ---------------------------------------------------------------------------
__global__ void gat_zero_kernel(float* __restrict__ p, int n) {
    int i = blockIdx.x * blockDim.x + threadIdx.x;
    int stride = gridDim.x * blockDim.x;
    for (; i < n; i += stride) p[i] = 0.0f;
}

// ---------------------------------------------------------------------------
// Scores: si = x . Wi^T, sj = x . Wj^T computed together as one 16-column
// GEMM using v_wmma_f32_16x16x32_f16. One wave per 16-row tile of x.
// B columns 0..7 = Wi heads, 8..15 = Wj heads.
// ---------------------------------------------------------------------------
__global__ __launch_bounds__(256) void gat_scores_kernel(
    const float* __restrict__ x, const float* __restrict__ Wi,
    const float* __restrict__ Wj, float* __restrict__ si,
    float* __restrict__ sj) {
    const int lane = threadIdx.x & 31;
    const int wave = threadIdx.x >> 5;
    const int tile = blockIdx.x * 8 + wave;
    if (tile >= N_TILES) return;           // uniform per wave: EXEC all-1s inside

    const int m0  = tile * 16;
    const int row = lane & 15;             // A-matrix row (M)
    const int hi  = lane >> 4;             // lane half selector
    const int n   = lane & 15;             // B/D column (N)

    // B column n: heads 0..7 -> Wi, 8..15 -> Wj
    const float* Wcol = (n < 8) ? (Wi + n * HIDDEN) : (Wj + (n - 8) * HIDDEN);

    v8f c = {};
    #pragma unroll
    for (int chunk = 0; chunk < 4; ++chunk) {
        const int koff = chunk * 32;
        v16h a, b;
        // A fragment: 16x32 tile of x (16-bit A layout, ISA 7.12.2)
        #pragma unroll
        for (int v = 0; v < 8; ++v) {
            const int K = koff + ((v < 4) ? 2 * v : 2 * v + 8) + hi * 8;
            const float* px = x + (m0 + row) * HIDDEN + K;
            a[2 * v]     = (_Float16)px[0];
            a[2 * v + 1] = (_Float16)px[1];
        }
        // B fragment: 32x16, same K order as A (table order), N = lane&15
        #pragma unroll
        for (int r = 0; r < 8; ++r) {
            const int K = koff + ((r < 4) ? 2 * r : 2 * r + 8) + hi * 8;
            b[2 * r]     = (_Float16)Wcol[K];
            b[2 * r + 1] = (_Float16)Wcol[K + 1];
        }
        c = __builtin_amdgcn_wmma_f32_16x16x32_f16(
                /*neg_a=*/false, a, /*neg_b=*/false, b,
                /*c_mod=*/(short)0, c, /*reuse_a=*/false, /*reuse_b=*/false);
    }

    // D layout: c[r] = D[M = r + 8*hi][N = lane&15]
    #pragma unroll
    for (int r = 0; r < 8; ++r) {
        const int rowD = m0 + r + 8 * hi;
        if (n < 8) si[rowD * HEADS + n]       = c[r];
        else       sj[rowD * HEADS + (n - 8)] = c[r];
    }
}

// ---------------------------------------------------------------------------
// Softmax denominators: denom[i][h] += exp(leaky_relu(si[i][h] + sj[j][h]))
// (max-shift omitted: logits are O(1), exp cannot overflow in f32; alpha is
//  shift-invariant so the result is mathematically identical)
// ---------------------------------------------------------------------------
__global__ void gat_denom_kernel(const int* __restrict__ ei,
                                 const float* __restrict__ si,
                                 const float* __restrict__ sj,
                                 float* __restrict__ denom) {
    const int e = blockIdx.x * blockDim.x + threadIdx.x;
    if (e >= N_EDGES) return;
    const int j = ei[e];            // source
    const int i = ei[N_EDGES + e];  // target
    const float* pi = si + i * HEADS;
    const float* pj = sj + j * HEADS;
    #pragma unroll
    for (int h = 0; h < HEADS; ++h) {
        float v = pi[h] + pj[h];
        v = (v > 0.0f) ? v : NEG_SLOPE * v;
        atomicAdd(&denom[i * HEADS + h], __expf(v));
    }
}

// ---------------------------------------------------------------------------
// Scatter: one wave per edge. Lanes 0..7 recompute the per-head normalized
// weights, wave-reduce to coef = mean_h(alpha_h), then each lane float4-loads
// x[j] and atomically accumulates coef * x[j] into out[i].
// ---------------------------------------------------------------------------
__global__ __launch_bounds__(256) void gat_scatter_kernel(
    const int* __restrict__ ei, const float* __restrict__ x,
    const float* __restrict__ si, const float* __restrict__ sj,
    const float* __restrict__ denom, float* __restrict__ out) {
    const int lane = threadIdx.x & 31;
    const int wave = threadIdx.x >> 5;
    const int e = blockIdx.x * 8 + wave;
    if (e >= N_EDGES) return;

    const int j = ei[e];
    const int i = ei[N_EDGES + e];

    const int h = lane & 7;
    float v = si[i * HEADS + h] + sj[j * HEADS + h];
    v = (v > 0.0f) ? v : NEG_SLOPE * v;
    float r = __expf(v) / denom[i * HEADS + h];
    r = (lane < 8) ? r : 0.0f;
    #pragma unroll
    for (int off = 16; off >= 1; off >>= 1)
        r += __shfl_xor(r, off, 32);
    const float coef = r * 0.125f;   // mean over 8 heads

    const float4 xv = *(const float4*)(x + (long)j * HIDDEN + lane * 4);
    float* po = out + (long)i * HIDDEN + lane * 4;
    atomicAdd(po + 0, coef * xv.x);
    atomicAdd(po + 1, coef * xv.y);
    atomicAdd(po + 2, coef * xv.z);
    atomicAdd(po + 3, coef * xv.w);
}

// ---------------------------------------------------------------------------
extern "C" void kernel_launch(void* const* d_in, const int* in_sizes, int n_in,
                              void* d_out, int out_size, void* d_ws, size_t ws_size,
                              hipStream_t stream) {
    const float* x  = (const float*)d_in[0];   // [N, 128]
    const float* Wi = (const float*)d_in[1];   // [8, 128]
    const float* Wj = (const float*)d_in[2];   // [8, 128]
    const int*   ei = (const int*)d_in[3];     // [2, E] (row0 = j/src, row1 = i/dst)
    float* out = (float*)d_out;                // [N, 128]

    float* ws    = (float*)d_ws;
    float* si    = ws;                          // [N, 8]
    float* sj    = ws + (size_t)N_NODES * HEADS;      // [N, 8]
    float* denom = ws + (size_t)2 * N_NODES * HEADS;  // [N, 8]

    gat_zero_kernel<<<2048, 256, 0, stream>>>(out, N_NODES * HIDDEN);
    gat_zero_kernel<<<512,  256, 0, stream>>>(denom, N_NODES * HEADS);

    gat_scores_kernel<<<(N_TILES + 7) / 8, 256, 0, stream>>>(x, Wi, Wj, si, sj);

    gat_denom_kernel<<<(N_EDGES + 255) / 256, 256, 0, stream>>>(ei, si, sj, denom);

    gat_scatter_kernel<<<(N_EDGES + 7) / 8, 256, 0, stream>>>(ei, x, si, sj, denom, out);
}
